// Gtu_44083544326435
// MI455X (gfx1250) — compile-verified, matrix operand order checked by reading, not run on hardware
//
#include <hip/hip_runtime.h>
#include <hip/hip_bf16.h>

// ---------------------------------------------------------------------------
// GTU pipeline for MI455X (gfx1250, wave32, WMMA, async global->LDS staging)
//   B=4 N=2048 E=1024 H=8 D1=3072 DH=384 R=512 GAMMA=0.99
// ---------------------------------------------------------------------------

#define B_   4
#define N_   2048
#define E_   1024
#define H_   8
#define D1_  3072
#define DH_  384
#define R_   512
#define M_   (B_ * N_)          // 8192 rows for the token GEMMs

typedef __attribute__((ext_vector_type(16))) __bf16 v16bf;
typedef __attribute__((ext_vector_type(8)))  float  v8f;
typedef __attribute__((ext_vector_type(4)))  unsigned int v4u;
typedef __attribute__((ext_vector_type(4)))  int v4i;

union Frag { v16bf v; v4u q[2]; };

__device__ __forceinline__ unsigned short bf16_rne(float f) {
    unsigned u = __float_as_uint(f);
    unsigned r = u + 0x7FFFu + ((u >> 16) & 1u);
    return (unsigned short)(r >> 16);
}

// ---- CDNA5 async global->LDS (guarded; falls back to sync copy) -----------
#if __has_builtin(__builtin_amdgcn_global_load_async_to_lds_b128)
#define HAVE_ASYNC_LDS 1
#else
#define HAVE_ASYNC_LDS 0
#endif

typedef __attribute__((address_space(1))) v4i GV4;   // global int4*
typedef __attribute__((address_space(3))) v4i LV4;   // LDS int4*

__device__ __forceinline__ void copy16_g2l(const unsigned short* g, unsigned short* l) {
#if HAVE_ASYNC_LDS
    __builtin_amdgcn_global_load_async_to_lds_b128((GV4*)(void*)g, (LV4*)(void*)l, 0, 0);
#else
    *(v4u*)l = *(const v4u*)g;
#endif
}

template <int CNT>
__device__ __forceinline__ void wait_async() {
#if HAVE_ASYNC_LDS
#if __has_builtin(__builtin_amdgcn_s_wait_asynccnt)
    __builtin_amdgcn_s_wait_asynccnt(CNT);
#else
    asm volatile("s_wait_asynccnt %0" :: "n"(CNT) : "memory");
#endif
#endif
}

// ---------------------------------------------------------------------------
// fp32 -> bf16 cast (grid-stride)
// ---------------------------------------------------------------------------
__global__ void cast_f32_bf16(const float* __restrict__ in,
                              unsigned short* __restrict__ out, int n) {
    int i = blockIdx.x * blockDim.x + threadIdx.x;
    int stride = gridDim.x * blockDim.x;
    for (; i < n; i += stride) out[i] = bf16_rne(in[i]);
}

// ---------------------------------------------------------------------------
// fp32 (K x N) -> bf16 transposed (N x K).  Tiled 32x32 via LDS, 256 threads.
// ---------------------------------------------------------------------------
__global__ void cast_transpose_bf16(const float* __restrict__ in,   // K x N
                                    unsigned short* __restrict__ out, // N x K
                                    int K, int N) {
    __shared__ float tile[32][33];
    const int nt = blockIdx.x * 32;
    const int kt = blockIdx.y * 32;
    const int tx = threadIdx.x & 31;
    const int ty = threadIdx.x >> 5;       // 0..7
#pragma unroll
    for (int r = ty; r < 32; r += 8)
        tile[r][tx] = in[(size_t)(kt + r) * N + nt + tx];
    __syncthreads();
#pragma unroll
    for (int r = ty; r < 32; r += 8)
        out[(size_t)(nt + r) * K + kt + tx] = bf16_rne(tile[tx][r]);
}

// ---------------------------------------------------------------------------
// bf16 WMMA GEMM: C[M,N] = act(A[M,K] @ Bt[N,K]^T + bias)
//   A row-major (M,K) bf16; Bt row-major (N,K) bf16 (pre-transposed weights).
//   act==1 -> SiLU.  M multiple of 128; N multiple of 256; K multiple of 32.
//   Block: 256 threads = 8 waves; block tile 128x256; wave tile 64x64.
//   Double-buffered async global->LDS staging (6 b128 asyncs per thread/tile).
//   B-fragments are consumed one at a time to keep VGPR pressure < 256.
// ---------------------------------------------------------------------------
#define LDT 40   // LDS row stride in ushorts (80B rows, 16B-aligned)
#define ATSZ (128 * LDT)    // one A tile buffer (128 rows x 32 k)
#define BTSZ (256 * LDT)    // one B tile buffer (256 cols x 32 k)

// copy ROWS x 32 bf16 tile: G[(r0+row)*ldg + k0 + ...] -> lbuf[row*LDT + ...]
template <int ROWS>
__device__ __forceinline__ void stage_tile(const unsigned short* __restrict__ G,
                                           int ldg, int r0, int k0,
                                           unsigned short* lbuf, int tid) {
#pragma unroll
    for (int i = 0; i < ROWS / 64; ++i) {   // ROWS*4 quads / 256 threads
        int q   = tid + 256 * i;
        int row = q >> 2;
        int qc  = q & 3;
        copy16_g2l(G + (size_t)(r0 + row) * ldg + k0 + qc * 8,
                   lbuf + row * LDT + qc * 8);
    }
}

__global__ void __launch_bounds__(256, 1)
gemm_bf16_wmma(const unsigned short* __restrict__ A,
               const unsigned short* __restrict__ Bt,
               const float* __restrict__ bias,
               float* __restrict__ C,
               int M, int N, int K, int act) {
    __shared__ __align__(16) unsigned short lsA[2 * ATSZ];  // 20 KB
    __shared__ __align__(16) unsigned short lsB[2 * BTSZ];  // 40 KB

    const int tid   = threadIdx.x;
    const int wave  = tid >> 5;
    const int lane  = tid & 31;
    const int waveM = wave >> 2;            // 0..1  (64-row half)
    const int waveN = wave & 3;             // 0..3  (64-col quarter)
    const int lrow  = lane & 15;
    const int kb    = (lane < 16) ? 0 : 8;  // per-lane K base (ISA 16-bit layout)
    const int m0 = blockIdx.y * 128;
    const int n0 = blockIdx.x * 256;

    // LDS fragment offsets (in ushorts), hoisted out of the K loop
    int aOff[4], bOff[4];
#pragma unroll
    for (int tm = 0; tm < 4; ++tm) aOff[tm] = (waveM * 64 + tm * 16 + lrow) * LDT + kb;
#pragma unroll
    for (int tn = 0; tn < 4; ++tn) bOff[tn] = (waveN * 64 + tn * 16 + lrow) * LDT + kb;

    v8f acc[16];
#pragma unroll
    for (int i = 0; i < 16; ++i) acc[i] = v8f{};

    const int NT = K >> 5;
    // prologue: stage tile 0 into buffer 0
    stage_tile<128>(A,  K, m0, 0, lsA, tid);
    stage_tile<256>(Bt, K, n0, 0, lsB, tid);

    for (int t = 0; t < NT; ++t) {
        if (t + 1 < NT) {
            int pb = (t + 1) & 1;
            stage_tile<128>(A,  K, m0, (t + 1) << 5, lsA + pb * ATSZ, tid);
            stage_tile<256>(Bt, K, n0, (t + 1) << 5, lsB + pb * BTSZ, tid);
            wait_async<6>();                // tile t landed; t+1's 6 in flight
        } else {
            wait_async<0>();
        }
        __syncthreads();

        const unsigned short* lA = lsA + (t & 1) * ATSZ;
        const unsigned short* lB = lsB + (t & 1) * BTSZ;

        // all 4 A fragments stay live (32 VGPRs)
        Frag aF[4];
#pragma unroll
        for (int tm = 0; tm < 4; ++tm) {
            aF[tm].q[0] = *(const v4u*)(lA + aOff[tm]);
            aF[tm].q[1] = *(const v4u*)(lA + aOff[tm] + 16);
        }
        // B fragments consumed one at a time (8 VGPRs live)
#pragma unroll
        for (int tn = 0; tn < 4; ++tn) {
            Frag bF;
            bF.q[0] = *(const v4u*)(lB + bOff[tn]);
            bF.q[1] = *(const v4u*)(lB + bOff[tn] + 16);
#pragma unroll
            for (int tm = 0; tm < 4; ++tm)
                acc[tm * 4 + tn] = __builtin_amdgcn_wmma_f32_16x16x32_bf16(
                    false, aF[tm].v, false, bF.v,
                    (short)0, acc[tm * 4 + tn], false, false);
        }
        __syncthreads();   // all reads done before next iter re-stages this buffer
    }

    // ---- epilogue: bias (+SiLU), fp32 store ----
#pragma unroll
    for (int tm = 0; tm < 4; ++tm) {
#pragma unroll
        for (int tn = 0; tn < 4; ++tn) {
            int col = n0 + waveN * 64 + tn * 16 + lrow;
            float bv = bias[col];
            int rbase = m0 + waveM * 64 + tm * 16 + ((lane >= 16) ? 8 : 0);
            v8f a = acc[tm * 4 + tn];
#pragma unroll
            for (int e = 0; e < 8; ++e) {
                float val = a[e] + bv;
                if (act) val = val / (1.0f + __expf(-val));   // SiLU
                C[(size_t)(rbase + e) * N + col] = val;
            }
        }
    }
}

// ---------------------------------------------------------------------------
// Position MLP
// ---------------------------------------------------------------------------
__global__ void pos_init_kernel(const float* __restrict__ pw,
                                const float* __restrict__ pb,
                                float* __restrict__ h0) {
    int i = blockIdx.x * 256 + threadIdx.x;          // over N_*R_
    if (i < N_ * R_) {
        int n = i >> 9;            // /512
        int r = i & (R_ - 1);
        h0[i] = (float)n * pw[r] + pb[r];
    }
}

// out[row] = relu(srms(in[row])) @ W + bias ; din = 512, dout in {512,3072}
__global__ void mlp_layer_kernel(const float* __restrict__ in,
                                 const float* __restrict__ W,
                                 const float* __restrict__ bias,
                                 float* __restrict__ out, int dout) {
    __shared__ float t[R_];
    __shared__ float red[256];
    const int row = blockIdx.x;
    const int tid = threadIdx.x;

    float local = 0.f;
    for (int r = tid; r < R_; r += 256) {
        float x = in[(size_t)row * R_ + r];
        t[r] = x;
        local += x * x;
    }
    red[tid] = local;
    __syncthreads();
    for (int s = 128; s > 0; s >>= 1) {
        if (tid < s) red[tid] += red[tid + s];
        __syncthreads();
    }
    float rms = sqrtf(red[0] * (1.0f / R_));
    float scale = 1.0f / (rms + 1e-8f);
    for (int r = tid; r < R_; r += 256) {
        float x = t[r] * scale;
        t[r] = x > 0.f ? x : 0.f;
    }
    __syncthreads();
    for (int c = tid; c < dout; c += 256) {
        float acc = bias[c];
        for (int r = 0; r < R_; ++r) acc = fmaf(t[r], W[(size_t)r * dout + c], acc);
        out[(size_t)row * dout + c] = acc;
    }
}

// ---------------------------------------------------------------------------
// FFT-based causal long convolution, fused with u * tv and bf16 cast.
// One block per (b,h,d) channel. Radix-2, length 4096, in 32KB LDS.
// DIF forward (bit-reversed out) + DIT inverse (bit-reversed in) => no reversal.
// ---------------------------------------------------------------------------
#define FFT_N 4096
#define PI_F 3.14159265358979323846f

__device__ __forceinline__ void fft4096_dif(float2* s, int tid) {
    for (int lh = 11; lh >= 0; --lh) {
        int half = 1 << lh;
        float w0 = -PI_F / (float)half;          // -2*pi/L, L = 2*half
        for (int k = tid; k < FFT_N / 2; k += 256) {
            int j  = k & (half - 1);
            int i0 = ((k >> lh) << (lh + 1)) + j;
            int i1 = i0 + half;
            float2 a = s[i0], b = s[i1];
            float dx = a.x - b.x, dy = a.y - b.y;
            float sn, cs;
            __sincosf(w0 * (float)j, &sn, &cs);
            s[i0] = make_float2(a.x + b.x, a.y + b.y);
            s[i1] = make_float2(dx * cs - dy * sn, dx * sn + dy * cs);
        }
        __syncthreads();
    }
}

__device__ __forceinline__ void fft4096_dit_inv(float2* s, int tid) {
    for (int lh = 0; lh <= 11; ++lh) {
        int half = 1 << lh;
        float w0 = PI_F / (float)half;           // +2*pi/L (conjugate twiddle)
        for (int k = tid; k < FFT_N / 2; k += 256) {
            int j  = k & (half - 1);
            int i0 = ((k >> lh) << (lh + 1)) + j;
            int i1 = i0 + half;
            float2 a = s[i0], b = s[i1];
            float sn, cs;
            __sincosf(w0 * (float)j, &sn, &cs);
            float tx = b.x * cs - b.y * sn;
            float ty = b.x * sn + b.y * cs;
            s[i0] = make_float2(a.x + tx, a.y + ty);
            s[i1] = make_float2(a.x - tx, a.y - ty);
        }
        __syncthreads();
    }
}

__global__ void conv_fft_kernel(const float* __restrict__ coefs,  // [N_, D1_]
                                const float* __restrict__ v,      // [M_, D1_]
                                const float* __restrict__ u,      // [M_, D1_]
                                unsigned short* __restrict__ yb)  // [M_, D1_] bf16
{
    __shared__ float2 s[FFT_N];      // 32 KB of the 320 KB WGP LDS
    const int tid = threadIdx.x;
    const int bid = blockIdx.x;      // B_*H_*DH_ = 12288
    const int d  = bid % DH_;
    const int h  = (bid / DH_) % H_;
    const int b  = bid / (DH_ * H_);
    const int ch = h * DH_ + d;
    const float lg = -0.010050335853501441f;     // logf(0.99)

    // ---- build a[k] = coefs[k]*gamma^k (a[0]=a[N]=coefs[0]), zero-padded ----
    for (int i = tid; i < FFT_N; i += 256) {
        float val = 0.f;
        if (i == 0 || i == N_)      val = coefs[ch];
        else if (i < N_)            val = coefs[(size_t)i * D1_ + ch] * __expf(lg * (float)i);
        s[i] = make_float2(val, 0.f);
    }
    __syncthreads();
    fft4096_dif(s, tid);

    float2 areg[16];
#pragma unroll
    for (int j = 0; j < 16; ++j) areg[j] = s[tid + 256 * j];
    // each thread only rewrites its own indices below -> no extra barrier needed

    // ---- v channel, zero-padded ----
    const float* vch = v + (size_t)b * N_ * D1_ + ch;
    for (int i = tid; i < FFT_N; i += 256) {
        float val = (i < N_) ? vch[(size_t)i * D1_] : 0.f;
        s[i] = make_float2(val, 0.f);
    }
    __syncthreads();
    fft4096_dif(s, tid);

    // ---- pointwise complex multiply (both spectra bit-reversed), 1/N fold ----
    const float inv = 1.0f / (float)FFT_N;
#pragma unroll
    for (int j = 0; j < 16; ++j) {
        float2 x = s[tid + 256 * j];
        float2 a = areg[j];
        s[tid + 256 * j] = make_float2((a.x * x.x - a.y * x.y) * inv,
                                       (a.x * x.y + a.y * x.x) * inv);
    }
    __syncthreads();
    fft4096_dit_inv(s, tid);

    // ---- tv[t] (t < N_), fuse u*tv, cast bf16 for output GEMM ----
    const float* uch = u + (size_t)b * N_ * D1_ + ch;
    unsigned short* ych = yb + (size_t)b * N_ * D1_ + ch;
    for (int i = tid; i < N_; i += 256) {
        float r = s[i].x * uch[(size_t)i * D1_];
        ych[(size_t)i * D1_] = bf16_rne(r);
    }
}

// ---------------------------------------------------------------------------
// Host-side launch
// ---------------------------------------------------------------------------
extern "C" void kernel_launch(void* const* d_in, const int* in_sizes, int n_in,
                              void* d_out, int out_size, void* d_ws, size_t ws_size,
                              hipStream_t stream) {
    const float* x     = (const float*)d_in[0];
    const float* u_w   = (const float*)d_in[1];
    const float* u_b   = (const float*)d_in[2];
    const float* v_w   = (const float*)d_in[3];
    const float* v_b   = (const float*)d_in[4];
    const float* o_w   = (const float*)d_in[5];
    const float* o_b   = (const float*)d_in[6];
    const float* pos_w = (const float*)d_in[7];
    const float* pos_b = (const float*)d_in[8];
    const float* lw0   = (const float*)d_in[9];
    const float* lb0   = (const float*)d_in[10];
    const float* lw1   = (const float*)d_in[11];
    const float* lb1   = (const float*)d_in[12];
    const float* lw2   = (const float*)d_in[13];
    const float* lb2   = (const float*)d_in[14];
    const float* out_w = (const float*)d_in[15];
    const float* out_b = (const float*)d_in[16];

    char* ws = (char*)d_ws;
    size_t off = 0;
    auto alloc = [&](size_t bytes) {
        char* p = ws + off;
        off += (bytes + 255) & ~(size_t)255;
        return p;
    };
    unsigned short* xb    = (unsigned short*)alloc((size_t)M_ * E_ * 2);    // 16 MB
    unsigned short* uwt   = (unsigned short*)alloc((size_t)E_ * D1_ * 2);   // 6 MB  (D1_ x E_)
    unsigned short* vwt   = (unsigned short*)alloc((size_t)E_ * D1_ * 2);   // 6 MB  (D1_ x E_)
    unsigned short* owt   = (unsigned short*)alloc((size_t)D1_ * E_ * 2);   // 6 MB  (E_ x D1_)
    float*          ubuf  = (float*)alloc((size_t)M_ * D1_ * 4);            // 96 MB
    float*          vbuf  = (float*)alloc((size_t)M_ * D1_ * 4);            // 96 MB
    float*          coefs = (float*)alloc((size_t)N_ * D1_ * 4);            // 24 MB
    float*          h0    = (float*)alloc((size_t)N_ * R_ * 4);             // 4 MB
    float*          h1    = (float*)alloc((size_t)N_ * R_ * 4);             // 4 MB
    unsigned short* yb    = (unsigned short*)alloc((size_t)M_ * D1_ * 2);   // 48 MB

    // 1) casts: x -> bf16; weights -> bf16 transposed (N,K) for b128 staging
    cast_f32_bf16<<<2048, 256, 0, stream>>>(x, xb, M_ * E_);
    {
        dim3 gT(D1_ / 32, E_ / 32);     // u_w, v_w: (E_ x D1_) -> (D1_ x E_)
        cast_transpose_bf16<<<gT, 256, 0, stream>>>(u_w, uwt, E_, D1_);
        cast_transpose_bf16<<<gT, 256, 0, stream>>>(v_w, vwt, E_, D1_);
        dim3 gO(E_ / 32, D1_ / 32);     // o_w: (D1_ x E_) -> (E_ x D1_)
        cast_transpose_bf16<<<gO, 256, 0, stream>>>(o_w, owt, D1_, E_);
    }

    // 2) u = silu(x @ u_w + u_b), v = silu(x @ v_w + v_b)
    dim3 gUV(D1_ / 256, M_ / 128);
    gemm_bf16_wmma<<<gUV, 256, 0, stream>>>(xb, uwt, u_b, ubuf, M_, D1_, E_, 1);
    gemm_bf16_wmma<<<gUV, 256, 0, stream>>>(xb, vwt, v_b, vbuf, M_, D1_, E_, 1);

    // 3) position MLP -> coefs
    pos_init_kernel<<<(N_ * R_ + 255) / 256, 256, 0, stream>>>(pos_w, pos_b, h0);
    mlp_layer_kernel<<<N_, 256, 0, stream>>>(h0, lw0, lb0, h1, R_);
    mlp_layer_kernel<<<N_, 256, 0, stream>>>(h1, lw1, lb1, h0, R_);
    mlp_layer_kernel<<<N_, 256, 0, stream>>>(h0, lw2, lb2, h1, R_);
    mlp_layer_kernel<<<N_, 256, 0, stream>>>(h1, out_w, out_b, coefs, D1_);

    // 4) FFT long-conv per channel, fused with u*tv, bf16 out
    conv_fft_kernel<<<B_ * H_ * DH_, 256, 0, stream>>>(coefs, vbuf, ubuf, yb);

    // 5) out = (u*tv) @ o_w + o_b   -> d_out (fp32)
    dim3 gG(E_ / 256, M_ / 128);
    gemm_bf16_wmma<<<gG, 256, 0, stream>>>(yb, owt, o_b, (float*)d_out, M_, E_, D1_, 0);
}
